// CrossAttention_34041910788264
// MI455X (gfx1250) — matrix operand order repaired and checked by reading, not compiled, not run on hardware
//
#include <hip/hip_runtime.h>
#include <hip/hip_bf16.h>
#include <math.h>

// ---------------------------------------------------------------------------
// CDNA5 / gfx1250 cross (channel) attention.
//   DIM=192, HEADS=8 (24 ch/head), B=2, H=W=256, HW=65536.
// GEMMs: v_wmma_f32_16x16x32_f16; panel staging: TDM tensor_load_to_lds with
// DMA-inserted LDS padding (16B per 128B -> 432B rows, conflict-free b128).
// ---------------------------------------------------------------------------

#define BN    2
#define CDIM  192
#define CKV   384
#define HEADS 8
#define CH    24
#define HH    256
#define WWID  256
#define HWSZ  65536

typedef __attribute__((ext_vector_type(16))) _Float16 v16h;
typedef __attribute__((ext_vector_type(8)))  float    v8f;
typedef unsigned int u32x4 __attribute__((ext_vector_type(4)));
typedef int          i32x4 __attribute__((ext_vector_type(4)));
typedef int          i32x8 __attribute__((ext_vector_type(8)));

union Frag16 { v16h v; float4 f4[2]; };
union AccF   { v8f  v; float  f[8];  };

__device__ __forceinline__ v8f wmma_f16(const Frag16& a, const Frag16& b, v8f c) {
  return __builtin_amdgcn_wmma_f32_16x16x32_f16(false, a.v, false, b.v,
                                                (short)0, c, false, false);
}

// Low 32 bits of a generic (flat) shared pointer == DS-relative LDS byte addr.
__device__ __forceinline__ unsigned lds_addr_of(const void* p) {
  return (unsigned)(unsigned long long)p;
}

// ---------------------------------------------------------------------------
// 1-D TDM copy: `halves` f16 elements from contiguous global memory to LDS,
// inserting 16B of LDS padding after every 128B (pad_interval code 4 = 32 dw,
// pad_amount code 3 = 4 dw).  D# layout per CDNA5 ISA 8.3-8.6:
//   g0: count=1 | lds_addr | global_addr[56:0] | type=2
//   g1: data_size=1(2B), pad_enable, pad_interval=4, pad_amount=3,
//       tensor_dim0 = tile_dim0 = halves, tensor_dim1 = 1, stride0 = halves
//   g2/g3: zero (tile_dim3/4 = 0 -> unused; <=2D tensor)
// This toolchain exposes the 6-operand builtin (extra i32x8 group; zero here).
// ---------------------------------------------------------------------------
__device__ __forceinline__ void tdm_load_1d_padded(const void* gsrc,
                                                   unsigned lds_byte_addr,
                                                   unsigned halves) {
  unsigned long long ga = (unsigned long long)gsrc;
  u32x4 g0;
  g0.x = 1u;                                           // count=1 user D#
  g0.y = lds_byte_addr;                                // LDS dest (bytes)
  g0.z = (unsigned)(ga & 0xFFFFFFFFu);                 // global_addr[31:0]
  g0.w = (unsigned)((ga >> 32) & 0x01FFFFFFu) | (2u << 30);  // [56:32] | type=2
  i32x8 g1 = {0, 0, 0, 0, 0, 0, 0, 0};
  g1[0] = (1 << 16)     // data_size = 2 bytes
        | (1 << 20)     // pad_enable
        | (4 << 22)     // pad_interval: 32 dwords (128B)
        | (3 << 25);    // pad_amount: 4 dwords (16B)
  g1[1] = (int)((halves & 0xFFFFu) << 16);             // tensor_dim0[15:0]
  g1[2] = (int)(((halves >> 16) & 0xFFFFu) | (1u << 16)); // dim0 hi | tensor_dim1=1
  g1[3] = (int)((halves & 0xFFFFu) << 16);             // tile_dim0
  g1[4] = 0;                                           // tile_dim1/2 = 0 (1-D)
  g1[5] = (int)halves;                                 // tensor_dim0_stride lo
  i32x4 g2 = {0, 0, 0, 0};
  i32x4 g3 = {0, 0, 0, 0};
  i32x8 g4 = {0, 0, 0, 0, 0, 0, 0, 0};
  __builtin_amdgcn_tensor_load_to_lds(g0, g1, g2, g3, g4, 0);
}

// Padded-LDS indexing: 192-half rows stored with +8 halves per 64 halves
// -> 216-half (432B) rows; stride 108 dwords mod 64 = 44 -> all 16 lanes of a
// fragment land on distinct 4-dword bank groups (conflict-free), 16B aligned.
#define ROWH 216
__device__ __forceinline__ int lds_idx(int r, int c) {
  return r * ROWH + c + 8 * (c >> 6);
}

// ---------------------------------------------------------------------------
// f32 -> f16 elementwise convert (weights)
// ---------------------------------------------------------------------------
__global__ __launch_bounds__(256) void k_cvt(const float* __restrict__ src,
                                             _Float16* __restrict__ dst, int n) {
  int i = blockIdx.x * 256 + threadIdx.x;
  if (i < n) dst[i] = (_Float16)src[i];
}

// ---------------------------------------------------------------------------
// Transpose [b][C][HW] f32  ->  [b][HW][C] f16  (n-major, 384B rows)
// ---------------------------------------------------------------------------
__global__ __launch_bounds__(256) void k_transpose(const float* __restrict__ x,
                                                   _Float16* __restrict__ xT) {
  __shared__ _Float16 T[64][200];
  const int b  = blockIdx.y;
  const int n0 = blockIdx.x * 64;
  for (int i = threadIdx.x; i < CDIM * 16; i += 256) {
    int c = i >> 4, ch = i & 15;
    float4 d = *(const float4*)(x + ((size_t)b * CDIM + c) * HWSZ + n0 + ch * 4);
    T[ch * 4 + 0][c] = (_Float16)d.x;
    T[ch * 4 + 1][c] = (_Float16)d.y;
    T[ch * 4 + 2][c] = (_Float16)d.z;
    T[ch * 4 + 3][c] = (_Float16)d.w;
  }
  __syncthreads();
  for (int i = threadIdx.x; i < 64 * 24; i += 256) {
    int r = i / 24, ch = i % 24;
    *(float4*)(xT + ((size_t)b * HWSZ + n0 + r) * CDIM + ch * 8) =
        *(const float4*)(&T[r][ch * 8]);
  }
}

// ---------------------------------------------------------------------------
// WMMA GEMM:  Y[b][co][n] = sum_ci W[co][ci] * XT[b][n][ci],  CI = 192.
// Block tile 64(CO) x 64(N); both panels are contiguous 24KB regions ->
// one 1-D TDM descriptor each, issued by wave 0, padded in-flight.
// ---------------------------------------------------------------------------
__global__ __launch_bounds__(256) void k_gemm(const _Float16* __restrict__ Wt,
                                              size_t w_bstride,
                                              const _Float16* __restrict__ XT,
                                              void* __restrict__ Y,
                                              int CO, int store_f32) {
  __shared__ __align__(16) _Float16 As[64 * ROWH];   // 27648 B
  __shared__ __align__(16) _Float16 Bs[64 * ROWH];   // 27648 B
  const int tid = threadIdx.x;
  const int n0  = blockIdx.x * 64;
  const int co0 = blockIdx.y * 64;
  const int b   = blockIdx.z;
  const _Float16* Wb = Wt + w_bstride * (size_t)b;

  if (tid < 32) {  // wave 0 drives the Tensor Data Mover
    tdm_load_1d_padded(Wb + (size_t)co0 * CDIM, lds_addr_of(As), 64 * CDIM);
    tdm_load_1d_padded(XT + ((size_t)b * HWSZ + n0) * CDIM, lds_addr_of(Bs),
                       64 * CDIM);
    __builtin_amdgcn_s_wait_tensorcnt(0);
  }
  __syncthreads();

  const int lane = tid & 31, w = tid >> 5;
  const int g = lane >> 4, l15 = lane & 15;
  const int mt  = w & 3;          // M-tile 0..3
  const int ntb = (w >> 2) * 2;   // N-tile base 0 or 2

  const int ar  = mt * 16 + l15;
  const int br0 = (ntb + 0) * 16 + l15;
  const int br1 = (ntb + 1) * 16 + l15;

  AccF acc0, acc1;
  acc0.v = v8f{0.f, 0.f, 0.f, 0.f, 0.f, 0.f, 0.f, 0.f};
  acc1.v = v8f{0.f, 0.f, 0.f, 0.f, 0.f, 0.f, 0.f, 0.f};

#pragma unroll
  for (int kc = 0; kc < CDIM; kc += 32) {
    Frag16 a, b0, b1;
    // A (16-bit 16x32): lane m+16g holds K=kc+16g..+15 of row m
    a.f4[0] = *(const float4*)(&As[lds_idx(ar, kc + 16 * g)]);
    a.f4[1] = *(const float4*)(&As[lds_idx(ar, kc + 16 * g + 8)]);
    // B (16-bit 32x16): lane n+16g holds K=kc+16g..+15 of column n (= XT row)
    const _Float16* pb0 = &Bs[lds_idx(br0, kc + 16 * g)];
    const _Float16* pb1 = &Bs[lds_idx(br1, kc + 16 * g)];
    b0.f4[0] = *(const float4*)(pb0);
    b0.f4[1] = *(const float4*)(pb0 + 8);
    b1.f4[0] = *(const float4*)(pb1);
    b1.f4[1] = *(const float4*)(pb1 + 8);
    acc0.v = wmma_f16(a, b0, acc0.v);
    acc1.v = wmma_f16(a, b1, acc1.v);
  }

  const size_t ybase = ((size_t)b * CO + co0) * (size_t)HWSZ;
  if (store_f32) {
    float* Yf = (float*)Y;
#pragma unroll
    for (int r = 0; r < 8; r++) {
      int m = mt * 16 + 8 * g + r;
      Yf[ybase + (size_t)m * HWSZ + n0 + (ntb + 0) * 16 + l15] = acc0.f[r];
      Yf[ybase + (size_t)m * HWSZ + n0 + (ntb + 1) * 16 + l15] = acc1.f[r];
    }
  } else {
    _Float16* Yh = (_Float16*)Y;
#pragma unroll
    for (int r = 0; r < 8; r++) {
      int m = mt * 16 + 8 * g + r;
      Yh[ybase + (size_t)m * HWSZ + n0 + (ntb + 0) * 16 + l15] = (_Float16)acc0.f[r];
      Yh[ybase + (size_t)m * HWSZ + n0 + (ntb + 1) * 16 + l15] = (_Float16)acc1.f[r];
    }
  }
}

// ---------------------------------------------------------------------------
// Depthwise 3x3 (SAME, zero pad), channel-major in/out, fused row sum-of-
// squares (for the L2 normalization) via block reduction + one atomic.
// ---------------------------------------------------------------------------
__global__ __launch_bounds__(256) void k_dwconv_c(const _Float16* __restrict__ in,
                                                  int Ctot, int c0,
                                                  const float* __restrict__ wdw, int w0,
                                                  _Float16* __restrict__ out,
                                                  float* __restrict__ ssq) {
  const int bc = blockIdx.x;
  const int b = bc / CDIM, c = bc % CDIM;
  const _Float16* ip = in + ((size_t)b * Ctot + c0 + c) * HWSZ;
  _Float16* op = out + ((size_t)b * CDIM + c) * HWSZ;
  float wv[9];
#pragma unroll
  for (int j = 0; j < 9; j++) wv[j] = wdw[(w0 + c) * 9 + j];

  const int p0 = blockIdx.y * 1024 + threadIdx.x * 4;
  float part = 0.f;
#pragma unroll
  for (int px = 0; px < 4; px++) {
    int p = p0 + px, y = p >> 8, x = p & 255;
    float acc = 0.f;
#pragma unroll
    for (int dy = -1; dy <= 1; dy++) {
      int yy = y + dy;
      if (yy < 0 || yy >= HH) continue;
#pragma unroll
      for (int dx = -1; dx <= 1; dx++) {
        int xx = x + dx;
        if (xx < 0 || xx >= WWID) continue;
        acc += wv[(dy + 1) * 3 + (dx + 1)] * (float)ip[yy * WWID + xx];
      }
    }
    op[p] = (_Float16)acc;
    part += acc * acc;
  }
  __shared__ float red[256];
  red[threadIdx.x] = part;
  __syncthreads();
  for (int s = 128; s > 0; s >>= 1) {
    if (threadIdx.x < s) red[threadIdx.x] += red[threadIdx.x + s];
    __syncthreads();
  }
  if (threadIdx.x == 0) atomicAdd(&ssq[b * CDIM + c], red[0]);
}

// ---------------------------------------------------------------------------
// Depthwise 3x3 for V, output written transposed [b][n][c] via LDS tile so the
// final WMMA GEMM gets contiguous n-major rows.
// ---------------------------------------------------------------------------
__global__ __launch_bounds__(256) void k_dwconv_vT(const _Float16* __restrict__ in,
                                                   const float* __restrict__ wdw,
                                                   _Float16* __restrict__ vT) {
  __shared__ _Float16 Os[64][200];
  const int b  = blockIdx.y;
  const int n0 = blockIdx.x * 64;
  const int y = n0 >> 8, x0 = n0 & 255;
  for (int i = threadIdx.x; i < CDIM * 64; i += 256) {
    int c = i >> 6, xl = i & 63;
    const _Float16* ip = in + ((size_t)b * CKV + CDIM + c) * HWSZ;
    const float* wv = wdw + (CDIM + c) * 9;
    int x = x0 + xl;
    float acc = 0.f;
#pragma unroll
    for (int dy = -1; dy <= 1; dy++) {
      int yy = y + dy;
      if (yy < 0 || yy >= HH) continue;
#pragma unroll
      for (int dx = -1; dx <= 1; dx++) {
        int xx = x + dx;
        if (xx < 0 || xx >= WWID) continue;
        acc += wv[(dy + 1) * 3 + (dx + 1)] * (float)ip[yy * WWID + xx];
      }
    }
    Os[xl][c] = (_Float16)acc;
  }
  __syncthreads();
  for (int i = threadIdx.x; i < 64 * 24; i += 256) {
    int r = i / 24, ch = i % 24;
    *(float4*)(vT + ((size_t)b * HWSZ + n0 + r) * CDIM + ch * 8) =
        *(const float4*)(&Os[r][ch * 8]);
  }
}

// ---------------------------------------------------------------------------
// Gram: S[bh][c][d] (padded 32x32) = sum_hw q[c,hw]*k[d,hw].
// Per (b,h): 2x2 WMMA 16x16x32 tiles (rows 24..31 zero-padded), each wave owns
// a 512-wide hw strip, partial tiles merged with f32 atomics.
// ---------------------------------------------------------------------------
__global__ __launch_bounds__(256) void k_gram(const _Float16* __restrict__ q,
                                              const _Float16* __restrict__ k,
                                              float* __restrict__ S) {
  const int lane = threadIdx.x & 31, w = threadIdx.x >> 5;
  const int g = lane >> 4, l15 = lane & 15;
  const int bh = blockIdx.y;
  const int b = bh >> 3, h = bh & 7;
  const size_t base = ((size_t)b * CDIM + h * CH) * (size_t)HWSZ;
  const int hw0 = blockIdx.x * 4096 + w * 512;

  const _Float16* qrow[2];
  const _Float16* krow[2];
  bool valid[2];
#pragma unroll
  for (int t = 0; t < 2; t++) {
    int r = t * 16 + l15;
    valid[t] = (r < CH);
    qrow[t] = q + base + (size_t)(valid[t] ? r : 0) * HWSZ;
    krow[t] = k + base + (size_t)(valid[t] ? r : 0) * HWSZ;
  }

  AccF acc[2][2];
#pragma unroll
  for (int i = 0; i < 2; i++)
#pragma unroll
    for (int j = 0; j < 2; j++)
      acc[i][j].v = v8f{0.f, 0.f, 0.f, 0.f, 0.f, 0.f, 0.f, 0.f};

  for (int hw = hw0; hw < hw0 + 512; hw += 32) {
    Frag16 a[2], bf[2];
#pragma unroll
    for (int t = 0; t < 2; t++) {
      if (valid[t]) {
        a[t].f4[0] = *(const float4*)(qrow[t] + hw + 16 * g);
        a[t].f4[1] = *(const float4*)(qrow[t] + hw + 16 * g + 8);
        bf[t].f4[0] = *(const float4*)(krow[t] + hw + 16 * g);
        bf[t].f4[1] = *(const float4*)(krow[t] + hw + 16 * g + 8);
      } else {
        a[t].f4[0] = make_float4(0.f, 0.f, 0.f, 0.f);
        a[t].f4[1] = make_float4(0.f, 0.f, 0.f, 0.f);
        bf[t].f4[0] = make_float4(0.f, 0.f, 0.f, 0.f);
        bf[t].f4[1] = make_float4(0.f, 0.f, 0.f, 0.f);
      }
    }
#pragma unroll
    for (int i = 0; i < 2; i++)
#pragma unroll
      for (int j = 0; j < 2; j++)
        acc[i][j].v = wmma_f16(a[i], bf[j], acc[i][j].v);
  }

  float* Sb = S + (size_t)bh * 1024;
#pragma unroll
  for (int i = 0; i < 2; i++)
#pragma unroll
    for (int j = 0; j < 2; j++)
#pragma unroll
      for (int r = 0; r < 8; r++) {
        int m = i * 16 + 8 * g + r;
        int n = j * 16 + l15;
        atomicAdd(&Sb[m * 32 + n], acc[i][j].f[r]);
      }
}

// ---------------------------------------------------------------------------
// Softmax with fused L2 normalization + temperature.
// ---------------------------------------------------------------------------
__global__ void k_softmax(const float* __restrict__ S,
                          const float* __restrict__ ssq_q,
                          const float* __restrict__ ssq_k,
                          const float* __restrict__ temp,
                          float* __restrict__ ATT) {
  const int bh = blockIdx.x;
  const int b = bh >> 3, h = bh & 7;
  const int c = threadIdx.x;
  if (c >= CH) return;
  const float* Sr = S + (size_t)bh * 1024 + c * 32;
  float nq = fmaxf(sqrtf(ssq_q[b * CDIM + h * CH + c]), 1e-12f);
  float t = temp[h];
  float v[CH];
  float mx = -3.0e38f;
  for (int d = 0; d < CH; d++) {
    float nk = fmaxf(sqrtf(ssq_k[b * CDIM + h * CH + d]), 1e-12f);
    v[d] = Sr[d] / (nq * nk) * t;
    mx = fmaxf(mx, v[d]);
  }
  float s = 0.f;
  for (int d = 0; d < CH; d++) { v[d] = __expf(v[d] - mx); s += v[d]; }
  float inv = 1.f / s;
  for (int d = 0; d < CH; d++) ATT[(size_t)bh * (CH * CH) + c * CH + d] = v[d] * inv;
}

// ---------------------------------------------------------------------------
// Fuse attn@V with the output projection: M[b][o][d] = sum_c wproj[o,h(d)*24+c]
// * attn[b,h(d),c,d%24]. Then Y = M_b @ V (one WMMA GEMM, no 96MB intermediate).
// ---------------------------------------------------------------------------
__global__ void k_buildM(const float* __restrict__ ATT,
                         const float* __restrict__ wproj,
                         _Float16* __restrict__ M16) {
  const int o = blockIdx.x, b = blockIdx.y, d = threadIdx.x;  // d < 192
  const int h = d / CH, dd = d % CH;
  const float* wp = wproj + o * CDIM + h * CH;
  const float* ar = ATT + ((size_t)(b * HEADS + h) * CH) * CH + dd;
  float acc = 0.f;
#pragma unroll
  for (int c = 0; c < CH; c++) acc += wp[c] * ar[c * CH];
  M16[((size_t)b * CDIM + o) * CDIM + d] = (_Float16)acc;
}

// ---------------------------------------------------------------------------
// Host orchestration (graph-capture safe: kernels + hipMemsetAsync only).
// ---------------------------------------------------------------------------
extern "C" void kernel_launch(void* const* d_in, const int* in_sizes, int n_in,
                              void* d_out, int out_size, void* d_ws, size_t ws_size,
                              hipStream_t stream) {
  (void)in_sizes; (void)n_in; (void)out_size; (void)ws_size;
  const float* x     = (const float*)d_in[0];
  const float* xref  = (const float*)d_in[1];
  const float* wq    = (const float*)d_in[2];
  const float* wqdw  = (const float*)d_in[3];
  const float* wkv   = (const float*)d_in[4];
  const float* wkvdw = (const float*)d_in[5];
  const float* wproj = (const float*)d_in[6];
  const float* temp  = (const float*)d_in[7];

  char* ws = (char*)d_ws;
  size_t off = 0;
  auto take = [&](size_t bytes) -> char* {
    char* p = ws + off;
    off = (off + bytes + 255) & ~(size_t)255;
    return p;
  };

  float*    S      = (float*)take(16 * 32 * 32 * sizeof(float));   // 65536 B
  float*    ssq_q  = (float*)take(BN * CDIM * sizeof(float));      // 1536 B
  float*    ssq_k  = (float*)take(BN * CDIM * sizeof(float));      // 1536 B
  float*    ATT    = (float*)take(16 * CH * CH * sizeof(float));
  _Float16* wq16   = (_Float16*)take((size_t)CDIM * CDIM * 2);
  _Float16* wkv16  = (_Float16*)take((size_t)CKV * CDIM * 2);
  _Float16* M16    = (_Float16*)take((size_t)BN * CDIM * CDIM * 2);

  const size_t big = (size_t)BN * HWSZ * CDIM * 2;  // 48 MiB
  _Float16* x16T  = (_Float16*)take(big);           // later aliased as q16
  _Float16* xr16T = (_Float16*)take(big);           // later aliased as k16
  _Float16* qpre  = (_Float16*)take(big);           // later aliased as vT
  _Float16* kvpre = (_Float16*)take(2 * big);       // 96 MiB

  _Float16* q16 = x16T;   // x16T dead after GEMM1
  _Float16* k16 = xr16T;  // xr16T dead after GEMM2
  _Float16* vT  = qpre;   // qpre dead after dwconv(q)

  (void)hipMemsetAsync(S, 0, 65536 + 1536 + 1536, stream);

  k_cvt<<<(CDIM * CDIM + 255) / 256, 256, 0, stream>>>(wq, wq16, CDIM * CDIM);
  k_cvt<<<(CKV * CDIM + 255) / 256, 256, 0, stream>>>(wkv, wkv16, CKV * CDIM);

  k_transpose<<<dim3(HWSZ / 64, BN), 256, 0, stream>>>(x, x16T);
  k_transpose<<<dim3(HWSZ / 64, BN), 256, 0, stream>>>(xref, xr16T);

  k_gemm<<<dim3(HWSZ / 64, CDIM / 64, BN), 256, 0, stream>>>(wq16, 0, x16T, qpre, CDIM, 0);
  k_gemm<<<dim3(HWSZ / 64, CKV / 64, BN), 256, 0, stream>>>(wkv16, 0, xr16T, kvpre, CKV, 0);

  k_dwconv_c<<<dim3(BN * CDIM, HWSZ / 1024), 256, 0, stream>>>(qpre, CDIM, 0, wqdw, 0, q16, ssq_q);
  k_dwconv_c<<<dim3(BN * CDIM, HWSZ / 1024), 256, 0, stream>>>(kvpre, CKV, 0, wkvdw, 0, k16, ssq_k);
  k_dwconv_vT<<<dim3(HWSZ / 64, BN), 256, 0, stream>>>(kvpre, wkvdw, vT);

  k_gram<<<dim3(16, BN * HEADS), 256, 0, stream>>>(q16, k16, S);
  k_softmax<<<BN * HEADS, 32, 0, stream>>>(S, ssq_q, ssq_k, temp, ATT);
  k_buildM<<<dim3(CDIM, BN), CDIM, 0, stream>>>(ATT, wproj, M16);

  k_gemm<<<dim3(HWSZ / 64, CDIM / 64, BN), 256, 0, stream>>>(
      M16, (size_t)CDIM * CDIM, vT, d_out, CDIM, 1);
}
